// AdapterBlock_89361089560703
// MI455X (gfx1250) — matrix operand order repaired and checked by reading, not compiled, not run on hardware
//
#include <hip/hip_runtime.h>

// ---------- types ----------
typedef __attribute__((ext_vector_type(16))) __bf16 v16bf;
typedef __attribute__((ext_vector_type(8)))  __bf16 v8bf;
typedef __attribute__((ext_vector_type(8)))  float  v8f;

#define DEV __device__ __forceinline__

static constexpr int   DIM   = 768;
static constexpr int   LH    = 2048;   // h tokens
static constexpr int   NV    = 1024;   // vis tokens
static constexpr int   SKV   = 3072;   // LH + NV
static constexpr int   BATCH = 4;
static constexpr int   HD    = 96;     // head dim
static constexpr float SCALE = 0.10206207261596577f; // 96^-0.5

DEV __bf16 f2bf(float f) { return (__bf16)f; }   // backend picks hw bf16 cvt

DEV v8f wmma_bf16(v16bf a, v16bf b, v8f c) {
    // D = A(16x32 bf16) * B(32x16 bf16) + C(16x16 f32)
    return __builtin_amdgcn_wmma_f32_16x16x32_bf16(
        false, a, false, b, (short)0, c, false, false);
}

DEV float rmax16(float v) {
    v = fmaxf(v, __shfl_xor(v, 1, 32));
    v = fmaxf(v, __shfl_xor(v, 2, 32));
    v = fmaxf(v, __shfl_xor(v, 4, 32));
    v = fmaxf(v, __shfl_xor(v, 8, 32));
    return v;
}
DEV float rsum16(float v) {
    v += __shfl_xor(v, 1, 32);
    v += __shfl_xor(v, 2, 32);
    v += __shfl_xor(v, 4, 32);
    v += __shfl_xor(v, 8, 32);
    return v;
}

// ============================================================
// Kernel 0: bulk f32 -> bf16 conversion (one-shot, out of hot loops)
// ============================================================
__global__ void cvt_kernel(const float* __restrict__ in, __bf16* __restrict__ out, int n8) {
    int i = blockIdx.x * 256 + threadIdx.x;
    if (i >= n8) return;
    const float4* p = (const float4*)(in + (size_t)i * 8);
    float4 a = p[0], b = p[1];
    v8bf v;
    v[0] = f2bf(a.x); v[1] = f2bf(a.y); v[2] = f2bf(a.z); v[3] = f2bf(a.w);
    v[4] = f2bf(b.x); v[5] = f2bf(b.y); v[6] = f2bf(b.z); v[7] = f2bf(b.w);
    *(v8bf*)(out + (size_t)i * 8) = v;
}

// Row r of the (possibly concatenated) bf16 activation matrix.
DEV const __bf16* src_row(const __bf16* A0, const __bf16* A1, int r, int concat) {
    if (!concat) return A0 + (size_t)r * DIM;
    int b = r / SKV, s = r - b * SKV;
    if (s < LH) return A0 + ((size_t)b * LH + s) * DIM;
    return          A1 + ((size_t)b * NV + (s - LH)) * DIM;
}

// ============================================================
// Unified GEMM:  out = A(bf16) @ W(bf16)^T
//   OUT_MODE 0: bf16 row-major               (Q/K paths)
//   OUT_MODE 1: bf16 transposed [b][col][s]  (V path)
//   OUT_MODE 2: f32 row-major                (Wo path)
// Block = 128 threads (4 waves); wave tile = 32(M) x 64(N).
// Inner K-step: 12x global_load_b128 -> 8x v_wmma (A reused 4x).
// ============================================================
template<int OUT_MODE>
__global__ void __launch_bounds__(128, 1)
gemm_bf16(const __bf16* __restrict__ A0, const __bf16* __restrict__ A1,
          const __bf16* __restrict__ W, void* __restrict__ outv,
          int concat, float out_scale) {
    const int tid  = threadIdx.x;
    const int wave = tid >> 5;
    const int lane = tid & 31;
    const int g    = lane >> 4;
    const int sub  = lane & 15;

    const int mbase = blockIdx.x * 64  + (wave & 1) * 32;
    const int nbase = blockIdx.y * 128 + (wave >> 1) * 64;

    const __bf16* pa[2];
    const __bf16* pw[4];
#pragma unroll
    for (int mi = 0; mi < 2; ++mi)
        pa[mi] = src_row(A0, A1, mbase + mi * 16 + sub, concat);
#pragma unroll
    for (int ni = 0; ni < 4; ++ni)
        pw[ni] = W + (size_t)(nbase + ni * 16 + sub) * DIM;

    v8f acc[2][4] = {};

    for (int c0 = 0; c0 < DIM; c0 += 32) {
        if (c0 + 64 < DIM) {   // prefetch ahead -> global_prefetch_b8
            __builtin_prefetch(pa[0] + c0 + 64, 0, 0);
            __builtin_prefetch(pw[0] + c0 + 64, 0, 0);
        }
        v16bf a[2], bb[4];
#pragma unroll
        for (int mi = 0; mi < 2; ++mi) {
            v8bf lo = *(const v8bf*)(pa[mi] + c0 + g * 16);
            v8bf hi = *(const v8bf*)(pa[mi] + c0 + g * 16 + 8);
#pragma unroll
            for (int j = 0; j < 8; ++j) { a[mi][j] = lo[j]; a[mi][8 + j] = hi[j]; }
        }
#pragma unroll
        for (int ni = 0; ni < 4; ++ni)
            bb[ni] = *(const v16bf*)(pw[ni] + c0 + g * 16);
#pragma unroll
        for (int mi = 0; mi < 2; ++mi)
#pragma unroll
            for (int ni = 0; ni < 4; ++ni)
                acc[mi][ni] = wmma_bf16(a[mi], bb[ni], acc[mi][ni]);
    }

#pragma unroll
    for (int mi = 0; mi < 2; ++mi)
#pragma unroll
        for (int ni = 0; ni < 4; ++ni)
#pragma unroll
            for (int i = 0; i < 8; ++i) {
                int r   = mbase + mi * 16 + i + 8 * g;
                int col = nbase + ni * 16 + sub;
                float v = acc[mi][ni][i] * out_scale;
                if constexpr (OUT_MODE == 0) {
                    ((__bf16*)outv)[(size_t)r * DIM + col] = f2bf(v);
                } else if constexpr (OUT_MODE == 1) {
                    int b = r / SKV, s = r - b * SKV;
                    ((__bf16*)outv)[((size_t)b * DIM + col) * SKV + s] = f2bf(v);
                } else {
                    ((float*)outv)[(size_t)r * DIM + col] = v;
                }
            }
}

// ============================================================
// Fused flash attention.
//   qb: [B*LH, DIM] bf16 (already scaled by SCALE)
//   kb: [B*SKV, DIM] bf16
//   vT: [B][DIM][SKV] bf16  (transposed V)
//   ao: [B*LH, DIM] bf16 out
// One wave handles 16 query rows of one head; 4 waves per block.
// All K then all V fragment loads are issued up front each tile:
// QK WMMAs wait only on K (in-order returns), V latency hides behind softmax.
// ============================================================
__global__ void __launch_bounds__(128, 1)
attn_kernel(const __bf16* __restrict__ qb, const __bf16* __restrict__ kb,
            const __bf16* __restrict__ vT, __bf16* __restrict__ ao) {
    const int tid  = threadIdx.x;
    const int wave = tid >> 5;
    const int lane = tid & 31;
    const int g    = lane >> 4;
    const int sub  = lane & 15;

    __shared__ __bf16 pbuf[4][16 * 40];   // padded stride 40 -> conflict-free frag reads
    __bf16* pb = &pbuf[wave][0];

    const int t   = blockIdx.x * 4 + wave;       // tile id in [0, 4096)
    const int b   = t >> 10;                     // 1024 tiles per batch
    const int rem = t & 1023;
    const int hh  = rem >> 7;                    // 128 tiles per head
    const int qt  = rem & 127;
    const size_t qrow0 = (size_t)b * LH + (size_t)qt * 16;

    // ---- load Q A-fragments (fixed over the whole KV loop) ----
    const __bf16* qp = qb + (qrow0 + sub) * DIM + hh * HD;
    v16bf aq[3];
#pragma unroll
    for (int dc = 0; dc < 3; ++dc) {
        v8bf lo = *(const v8bf*)(qp + dc * 32 + g * 16);
        v8bf hi = *(const v8bf*)(qp + dc * 32 + g * 16 + 8);
#pragma unroll
        for (int j = 0; j < 8; ++j) { aq[dc][j] = lo[j]; aq[dc][8 + j] = hi[j]; }
    }

    const __bf16* kbase = kb + (size_t)b * SKV * DIM + hh * HD;
    const __bf16* vbase = vT + ((size_t)b * DIM + hh * HD) * SKV;

    v8f acc[6] = {};
    float rm[8], rs[8];
#pragma unroll
    for (int i = 0; i < 8; ++i) { rm[i] = -3.0e38f; rs[i] = 0.0f; }

    for (int s0 = 0; s0 < SKV; s0 += 32) {
        if (s0 + 32 < SKV) {   // prefetch next K/V tiles
            __builtin_prefetch(kbase + (size_t)(s0 + 32 + sub) * DIM, 0, 0);
            __builtin_prefetch(vbase + (size_t)sub * SKV + s0 + 32, 0, 0);
        }

        // ---- issue ALL K fragment loads, then ALL V fragment loads ----
        const __bf16* kt = kbase + (size_t)s0 * DIM;
        v16bf bk[3][2];
#pragma unroll
        for (int dc = 0; dc < 3; ++dc) {
            bk[dc][0] = *(const v16bf*)(kt + (size_t)sub        * DIM + dc * 32 + g * 16);
            bk[dc][1] = *(const v16bf*)(kt + (size_t)(16 + sub) * DIM + dc * 32 + g * 16);
        }
        v16bf bv[6];
#pragma unroll
        for (int nf = 0; nf < 6; ++nf)
            bv[nf] = *(const v16bf*)(vbase + (size_t)(nf * 16 + sub) * SKV + s0 + g * 16);

        // ---- scores: QK^T over a 16(q) x 32(keys) tile (V still in flight) ----
        v8f sc0 = {}, sc1 = {};
#pragma unroll
        for (int dc = 0; dc < 3; ++dc) {
            sc0 = wmma_bf16(aq[dc], bk[dc][0], sc0);
            sc1 = wmma_bf16(aq[dc], bk[dc][1], sc1);
        }

        // ---- online softmax (row stats live across 16-lane halves) ----
        float pscale[8];
#pragma unroll
        for (int i = 0; i < 8; ++i) {
            float m  = rmax16(fmaxf(sc0[i], sc1[i]));
            float mn = fmaxf(rm[i], m);
            float sf = __expf(rm[i] - mn);
            rm[i] = mn;
            float p0 = __expf(sc0[i] - mn);
            float p1 = __expf(sc1[i] - mn);
            sc0[i] = p0; sc1[i] = p1;
            rs[i] = rs[i] * sf + rsum16(p0 + p1);
            pscale[i] = sf;
        }
#pragma unroll
        for (int nf = 0; nf < 6; ++nf)
#pragma unroll
            for (int i = 0; i < 8; ++i) acc[nf][i] *= pscale[i];

        // ---- P: C-layout -> LDS -> A-layout (bf16) ----
#pragma unroll
        for (int i = 0; i < 8; ++i) {
            int r = i + 8 * g;
            pb[r * 40 + sub]      = f2bf(sc0[i]);
            pb[r * 40 + 16 + sub] = f2bf(sc1[i]);
        }
        __builtin_amdgcn_wave_barrier();   // keep LDS store->load order
        v16bf ap;
        {
            v8bf lo = *(const v8bf*)(pb + sub * 40 + g * 16);
            v8bf hi = *(const v8bf*)(pb + sub * 40 + g * 16 + 8);
#pragma unroll
            for (int j = 0; j < 8; ++j) { ap[j] = lo[j]; ap[8 + j] = hi[j]; }
        }
        __builtin_amdgcn_wave_barrier();

        // ---- accumulate P @ V (fragments already resident) ----
#pragma unroll
        for (int nf = 0; nf < 6; ++nf)
            acc[nf] = wmma_bf16(ap, bv[nf], acc[nf]);
    }

    // ---- epilogue: divide by row sums, store bf16 ----
#pragma unroll
    for (int nf = 0; nf < 6; ++nf)
#pragma unroll
        for (int i = 0; i < 8; ++i) {
            float v = acc[nf][i] / rs[i];
            size_t row = qrow0 + i + 8 * g;
            ao[row * DIM + hh * HD + nf * 16 + sub] = f2bf(v);
        }
}

// ============================================================
// LayerNorm over last dim (768), one row per block.
// ============================================================
__global__ void ln_kernel(const float* __restrict__ x, const float* __restrict__ gamma,
                          const float* __restrict__ beta, float* __restrict__ out) {
    const int row = blockIdx.x;
    const int tid = threadIdx.x;
    __shared__ float s1[256], s2[256];

    const float* xr = x + (size_t)row * DIM;
    float s = 0.f, q = 0.f;
    for (int j = tid; j < DIM; j += 256) {
        float v = xr[j];
        s += v; q += v * v;
    }
    s1[tid] = s; s2[tid] = q;
    __syncthreads();
    for (int off = 128; off > 0; off >>= 1) {
        if (tid < off) { s1[tid] += s1[tid + off]; s2[tid] += s2[tid + off]; }
        __syncthreads();
    }
    const float mean = s1[0] * (1.0f / DIM);
    const float var  = s2[0] * (1.0f / DIM) - mean * mean;
    const float inv  = rsqrtf(var + 1e-5f);
    for (int j = tid; j < DIM; j += 256) {
        float v = xr[j];
        out[(size_t)row * DIM + j] = (v - mean) * inv * gamma[j] + beta[j];
    }
}

// ============================================================
// Host launcher
// ============================================================
extern "C" void kernel_launch(void* const* d_in, const int* in_sizes, int n_in,
                              void* d_out, int out_size, void* d_ws, size_t ws_size,
                              hipStream_t stream) {
    const float* h     = (const float*)d_in[0];
    const float* vis   = (const float*)d_in[1];
    const float* Wq    = (const float*)d_in[2];
    const float* Wk    = (const float*)d_in[3];
    const float* Wv    = (const float*)d_in[4];
    const float* Wo    = (const float*)d_in[5];
    const float* gamma = (const float*)d_in[6];
    const float* beta  = (const float*)d_in[7];
    float* outp = (float*)d_out;

    char* ws = (char*)d_ws;
    // ---- workspace layout (bytes), with lifetime-based overlays ----
    const size_t HB_OFF  = 0;                                        // h bf16   12,582,912
    const size_t VIS_OFF = HB_OFF  + (size_t)BATCH * LH * DIM * 2;   // vis bf16  6,291,456
    const size_t WQ_OFF  = VIS_OFF + (size_t)BATCH * NV * DIM * 2;   // 4 weights bf16
    const size_t WK_OFF  = WQ_OFF + (size_t)DIM * DIM * 2;
    const size_t WV_OFF  = WK_OFF + (size_t)DIM * DIM * 2;
    const size_t WO_OFF  = WV_OFF + (size_t)DIM * DIM * 2;
    const size_t QB_OFF  = WO_OFF + (size_t)DIM * DIM * 2;           // Q bf16   12,582,912
    const size_t KB_OFF  = QB_OFF + (size_t)BATCH * LH  * DIM * 2;   // K bf16   18,874,368
    const size_t VT_OFF  = KB_OFF + (size_t)BATCH * SKV * DIM * 2;   // V^T bf16 18,874,368
    const size_t AO_OFF  = 0;       // overlays dead h/vis bf16 copies (12.6MB <= 18.9MB)
    const size_t O2_OFF  = QB_OFF;  // f32, overlays dead Q + part of K (25.2MB <= 31.5MB)

    __bf16* hb   = (__bf16*)(ws + HB_OFF);
    __bf16* visb = (__bf16*)(ws + VIS_OFF);
    __bf16* wqb  = (__bf16*)(ws + WQ_OFF);
    __bf16* wkb  = (__bf16*)(ws + WK_OFF);
    __bf16* wvb  = (__bf16*)(ws + WV_OFF);
    __bf16* wob  = (__bf16*)(ws + WO_OFF);
    __bf16* qb   = (__bf16*)(ws + QB_OFF);
    __bf16* kb   = (__bf16*)(ws + KB_OFF);
    __bf16* vT   = (__bf16*)(ws + VT_OFF);
    __bf16* ao   = (__bf16*)(ws + AO_OFF);
    float*  o2   = (float*)(ws + O2_OFF);

    // ---- stage 0: one-shot f32 -> bf16 conversions ----
    const int nH = BATCH * LH * DIM / 8, nV = BATCH * NV * DIM / 8, nW = DIM * DIM / 8;
    cvt_kernel<<<dim3((nH + 255) / 256), dim3(256), 0, stream>>>(h,   hb,   nH);
    cvt_kernel<<<dim3((nV + 255) / 256), dim3(256), 0, stream>>>(vis, visb, nV);
    cvt_kernel<<<dim3((nW + 255) / 256), dim3(256), 0, stream>>>(Wq,  wqb,  nW);
    cvt_kernel<<<dim3((nW + 255) / 256), dim3(256), 0, stream>>>(Wk,  wkb,  nW);
    cvt_kernel<<<dim3((nW + 255) / 256), dim3(256), 0, stream>>>(Wv,  wvb,  nW);
    cvt_kernel<<<dim3((nW + 255) / 256), dim3(256), 0, stream>>>(Wo,  wob,  nW);

    dim3 blk(128);
    // ---- stage 1: projections (all WMMA); block tile 64(M) x 128(N) ----
    gemm_bf16<0><<<dim3(8192 / 64,  DIM / 128), blk, 0, stream>>>(hb, visb, wqb, qb, 0, SCALE);
    gemm_bf16<0><<<dim3(12288 / 64, DIM / 128), blk, 0, stream>>>(hb, visb, wkb, kb, 1, 1.0f);
    gemm_bf16<1><<<dim3(12288 / 64, DIM / 128), blk, 0, stream>>>(hb, visb, wvb, vT, 1, 1.0f);
    // ---- stage 2: flash attention (WMMA QK^T and PV) ----
    attn_kernel<<<dim3(1024), blk, 0, stream>>>(qb, kb, vT, ao);
    // ---- stage 3: output projection (WMMA) ----
    gemm_bf16<2><<<dim3(8192 / 64, DIM / 128), blk, 0, stream>>>(ao, nullptr, wob, o2, 0, 1.0f);
    // ---- stage 4: layernorm -> final output ----
    ln_kernel<<<dim3(BATCH * LH), dim3(256), 0, stream>>>(o2, gamma, beta, outp);
}
